// SparseMoEFFN_6803228197409
// MI455X (gfx1250) — compile-verified, matrix operand order checked by reading, not run on hardware
//
#include <hip/hip_runtime.h>
#include <cstdint>
#include <cstddef>

// Problem constants (match reference setup_inputs)
#define D_DIM 1024
#define F_DIM 4096
#define E_NUM 8
#define N_TOK 8192    // B*T = 4*2048
#define TILE_M 64     // tokens per workgroup tile
#define NWAVE 16      // 512 threads
#define FC 64         // F-chunk (K of second GEMM per iteration)
#define XROW 516      // LDS row stride in u32 for x tile (512 data + 4 pad)
#define HROW 36       // LDS row stride in u32 for h chunk (32 data + 4 pad)

typedef __attribute__((ext_vector_type(16))) __bf16 v16bf;
typedef __attribute__((ext_vector_type(8)))  float  v8f;
typedef __attribute__((ext_vector_type(8)))  unsigned int v8u;

__device__ __forceinline__ unsigned short f32_to_bf16(float f) {
  unsigned int u = __builtin_bit_cast(unsigned int, f);
  u += 0x7FFFu + ((u >> 16) & 1u);   // round-to-nearest-even
  return (unsigned short)(u >> 16);
}
__device__ __forceinline__ unsigned int pack_bf16x2(float lo, float hi) {
  return (unsigned int)f32_to_bf16(lo) | ((unsigned int)f32_to_bf16(hi) << 16);
}
__device__ __forceinline__ float gelu_tanh(float x) {
  const float c = 0.7978845608028654f; // sqrt(2/pi)
  float t = tanhf(c * (x + 0.044715f * x * x * x));
  return 0.5f * x * (1.0f + t);
}
// 16-bit A/B matrix K packing per CDNA5 ISA 7.12.2:
// VGPR j (j<4): K = 2j,2j+1 (lanes 0-15) / +8 (lanes 16-31); j>=4: +16.
__device__ __forceinline__ int kbase(int j, int hi) {
  return 2 * (j & 3) + ((j >> 2) << 4) + (hi << 3);
}
// Inverse: local K (0..31) -> (j, hi, t) position in the operand
__device__ __forceinline__ void kinv(int kl, int& j, int& hi, int& t) {
  t  = kl & 1;
  hi = (kl >> 3) & 1;
  j  = ((kl >> 1) & 3) | (((kl >> 4) & 1) << 2);
}

// ---------------------------------------------------------------------------
// Kernel 0: zero y and workspace header (counts[8], dens[8], probsum[8])
// ---------------------------------------------------------------------------
__global__ void moe_zero_kernel(float* __restrict__ y, unsigned int* __restrict__ hdr) {
  size_t idx = (size_t)blockIdx.x * blockDim.x + threadIdx.x;
  size_t total = (size_t)N_TOK * D_DIM;
  if (idx < total) y[idx] = 0.0f;
  if (blockIdx.x == 0 && threadIdx.x < 32) hdr[threadIdx.x] = 0u;
}

// ---------------------------------------------------------------------------
// Kernel 0b: pack fp32 weight pairs (k, k+1) along the leading dim into one
// u32 of two bf16. out layout: [rows/2][inner] u32, in: [rows][inner] f32.
// ---------------------------------------------------------------------------
__global__ void moe_pack_kernel(const float* __restrict__ in,
                                unsigned int* __restrict__ out,
                                int npairs, int inner) {
  int idx = blockIdx.x * blockDim.x + threadIdx.x;
  if (idx >= npairs) return;
  int f = idx % inner;
  int rest = idx / inner;
  size_t i0 = (size_t)rest * 2 * inner + f;
  out[idx] = pack_bf16x2(in[i0], in[i0 + inner]);
}

// ---------------------------------------------------------------------------
// Kernel 1: router. One wave32 per token: logits -> softmax -> top-2 ->
// renormalize; scatter into per-expert buckets; aux-loss statistics.
// ---------------------------------------------------------------------------
__global__ __launch_bounds__(256) void moe_router_kernel(
    const float* __restrict__ xf, const float* __restrict__ wr,
    int* __restrict__ counts, int* __restrict__ dens, float* __restrict__ probsum,
    int* __restrict__ bucket, float* __restrict__ gwt) {
  int wave = threadIdx.x >> 5;
  int lane = threadIdx.x & 31;
  int tok = blockIdx.x * 8 + wave;
  if (tok >= N_TOK) return;

  const float* xr = xf + (size_t)tok * D_DIM;
  float acc[E_NUM];
#pragma unroll
  for (int e = 0; e < E_NUM; ++e) acc[e] = 0.0f;
  for (int d = lane; d < D_DIM; d += 32) {
    float xv = xr[d];
    const float* w = wr + (size_t)d * E_NUM;  // w_router is [D, E] row-major
#pragma unroll
    for (int e = 0; e < E_NUM; ++e) acc[e] += xv * w[e];
  }
#pragma unroll
  for (int e = 0; e < E_NUM; ++e) {
#pragma unroll
    for (int m = 16; m >= 1; m >>= 1) acc[e] += __shfl_xor(acc[e], m, 32);
  }
  if (lane == 0) {
    float mx = acc[0];
#pragma unroll
    for (int e = 1; e < E_NUM; ++e) mx = fmaxf(mx, acc[e]);
    float p[E_NUM], s = 0.0f;
#pragma unroll
    for (int e = 0; e < E_NUM; ++e) { p[e] = __expf(acc[e] - mx); s += p[e]; }
    float inv = 1.0f / s;
#pragma unroll
    for (int e = 0; e < E_NUM; ++e) p[e] *= inv;
    // top-2 (distinct experts)
    int e0 = 0;
#pragma unroll
    for (int e = 1; e < E_NUM; ++e) if (p[e] > p[e0]) e0 = e;
    int e1 = (e0 == 0) ? 1 : 0;
#pragma unroll
    for (int e = 0; e < E_NUM; ++e) if (e != e0 && p[e] > p[e1]) e1 = e;
    float w0 = p[e0], w1 = p[e1];
    float wsum = w0 + w1;
    w0 /= wsum; w1 /= wsum;
    int pos0 = atomicAdd(&counts[e0], 1);
    bucket[e0 * N_TOK + pos0] = tok;  gwt[e0 * N_TOK + pos0] = w0;
    int pos1 = atomicAdd(&counts[e1], 1);
    bucket[e1 * N_TOK + pos1] = tok;  gwt[e1 * N_TOK + pos1] = w1;
    atomicAdd(&dens[e0], 1);
#pragma unroll
    for (int e = 0; e < E_NUM; ++e) atomicAdd(&probsum[e], p[e]);
  }
}

// ---------------------------------------------------------------------------
// Kernel 2: expert FFN tile. Workgroup = 16 wave32 = (expert, 64-token tile).
// Full [64 x 1024] output tile accumulated in registers (acc[4][4] v8f per
// wave) over 64 F-chunks. LDS holds x and gelu(h) pre-swizzled into WMMA
// A-operand order (8 consecutive dwords per operand -> ds_load_b128).
// PACKED: weights pre-converted to pair-packed bf16 in workspace (u32 loads,
// zero conversion VALU, 134 MB set is L2-resident). !PACKED: inline convert.
// ---------------------------------------------------------------------------
template <bool PACKED>
__global__ __launch_bounds__(512) void moe_ffn_kernel(
    const float* __restrict__ xf,
    const float* __restrict__ w1f, const unsigned int* __restrict__ w1p,
    const float* __restrict__ b1,
    const float* __restrict__ w2f, const unsigned int* __restrict__ w2p,
    const float* __restrict__ b2,
    const int* __restrict__ counts, const int* __restrict__ bucket,
    const float* __restrict__ gwt, float* __restrict__ y) {
  const int e = blockIdx.y;
  const int cnt = counts[e];
  const int base = blockIdx.x * TILE_M;
  if (base >= cnt) return;  // ragged grid: inactive tiles exit

  __shared__ unsigned int xs[TILE_M * XROW];  // x tile, bf16 pairs, A-swizzled
  __shared__ unsigned int hs[TILE_M * HROW];  // gelu(h) chunk, A-swizzled
  __shared__ int   toks[TILE_M];
  __shared__ float gws[TILE_M];

  const int tid  = threadIdx.x;
  const int wave = tid >> 5;
  const int lane = tid & 31;
  const int hi   = lane >> 4;   // lane half selects K+8 region
  const int lx   = lane & 15;

  if (tid < TILE_M) {
    int idx = base + tid;
    bool v = idx < cnt;
    toks[tid] = v ? bucket[e * N_TOK + idx] : bucket[e * N_TOK + base];
    gws[tid]  = v ? gwt[e * N_TOK + idx] : 0.0f;
  }
  __syncthreads();

  // Stage x tile into LDS as bf16, pre-swizzled into A-operand order
  for (int i = tid; i < TILE_M * D_DIM; i += 512) {
    int r = i >> 10, c = i & (D_DIM - 1);
    int kblk = c >> 5, kl = c & 31, j, hh, tt;
    kinv(kl, j, hh, tt);
    unsigned short v = f32_to_bf16(xf[(size_t)toks[r] * D_DIM + c]);
    ((unsigned short*)xs)[(r * XROW + kblk * 16 + hh * 8 + j) * 2 + tt] = v;
  }
  __syncthreads();

  // Output accumulators: 4 M-tiles x 4 N-tiles per wave
  // (wave owns d-columns [wave*64, wave*64+64)).
  v8f acc[4][4];
#pragma unroll
  for (int m = 0; m < 4; ++m)
#pragma unroll
    for (int t = 0; t < 4; ++t) acc[m][t] = v8f{0.f,0.f,0.f,0.f,0.f,0.f,0.f,0.f};

  const int mt1 = wave & 3;   // stage-1: M-tile of this wave (4)
  const int ft1 = wave >> 2;  // stage-1: F-subtile of this wave (4)

  for (int fc0 = 0; fc0 < F_DIM; fc0 += FC) {
    if (fc0 + FC < F_DIM) {
      if (PACKED)
        __builtin_prefetch(&w1p[((size_t)e * (D_DIM / 2) + (lane << 3)) * F_DIM + fc0 + FC], 0, 1);
      else
        __builtin_prefetch(&w1f[((size_t)e * D_DIM + (lane << 3)) * F_DIM + fc0 + FC], 0, 1);
    }

    // ---- Stage 1: h[64, FC] = gelu(x[64,1024] @ W1[1024, FC] + b1) ----
    v8f hacc = v8f{0.f,0.f,0.f,0.f,0.f,0.f,0.f,0.f};
    const int arow = mt1 * 16 + lx;
    const int fcol = fc0 + ft1 * 16 + lx;
    for (int ks = 0; ks < D_DIM; ks += 32) {
      v8u av, bv;
      const int abase = arow * XROW + (ks >> 5) * 16 + hi * 8;
#pragma unroll
      for (int j = 0; j < 8; ++j) av[j] = xs[abase + j];  // 8 consec dwords
#pragma unroll
      for (int j = 0; j < 8; ++j) {
        int kb = kbase(j, hi);
        if constexpr (PACKED) {
          bv[j] = w1p[((size_t)e * (D_DIM / 2) + ((ks + kb) >> 1)) * F_DIM + fcol];
        } else {
          const float* wp = &w1f[((size_t)e * D_DIM + ks + kb) * F_DIM + fcol];
          bv[j] = pack_bf16x2(wp[0], wp[F_DIM]);
        }
      }
      hacc = __builtin_amdgcn_wmma_f32_16x16x32_bf16(
          false, __builtin_bit_cast(v16bf, av),
          false, __builtin_bit_cast(v16bf, bv),
          (short)0, hacc, false, false);
    }
    __syncthreads();  // previous chunk's stage-2 reads of hs complete
#pragma unroll
    for (int r = 0; r < 8; ++r) {
      int row = mt1 * 16 + r + (hi << 3);
      int lcol = ft1 * 16 + lx;                 // K index within chunk
      float v = hacc[r] + b1[(size_t)e * F_DIM + fc0 + lcol];
      unsigned short hb = f32_to_bf16(gelu_tanh(v));
      int kblk = lcol >> 5, kl = lcol & 31, j, hh, tt;
      kinv(kl, j, hh, tt);
      ((unsigned short*)hs)[(row * HROW + kblk * 16 + hh * 8 + j) * 2 + tt] = hb;
    }
    __syncthreads();  // hs visible to all waves

    // ---- Stage 2: out[64,1024] += h[64,FC] @ W2[FC,1024] ----
#pragma unroll
    for (int kk = 0; kk < FC; kk += 32) {
      v16bf A[4];
#pragma unroll
      for (int m = 0; m < 4; ++m) {
        v8u a;
        const int hbase = (m * 16 + lx) * HROW + (kk >> 5) * 16 + hi * 8;
#pragma unroll
        for (int j = 0; j < 8; ++j) a[j] = hs[hbase + j];
        A[m] = __builtin_bit_cast(v16bf, a);
      }
#pragma unroll
      for (int t = 0; t < 4; ++t) {
        int dcol = (wave * 4 + t) * 16 + lx;
        v8u bv;
#pragma unroll
        for (int j = 0; j < 8; ++j) {
          int kf = fc0 + kk + kbase(j, hi);
          if constexpr (PACKED) {
            bv[j] = w2p[((size_t)e * (F_DIM / 2) + (kf >> 1)) * D_DIM + dcol];
          } else {
            const float* wp = &w2f[((size_t)e * F_DIM + kf) * D_DIM + dcol];
            bv[j] = pack_bf16x2(wp[0], wp[D_DIM]);
          }
        }
        v16bf B = __builtin_bit_cast(v16bf, bv);
#pragma unroll
        for (int m = 0; m < 4; ++m) {
          acc[m][t] = __builtin_amdgcn_wmma_f32_16x16x32_bf16(
              false, A[m], false, B, (short)0, acc[m][t], false, false);
        }
      }
    }
  }
  __syncthreads();

  // Combine: y[tok] += gate_w * (acc + b2), atomic f32 add (2 contributions
  // per token total across the two routed experts).
#pragma unroll
  for (int m = 0; m < 4; ++m) {
#pragma unroll
    for (int t = 0; t < 4; ++t) {
      int dcol = (wave * 4 + t) * 16 + lx;
      float bias = b2[(size_t)e * D_DIM + dcol];
#pragma unroll
      for (int r = 0; r < 8; ++r) {
        int row = m * 16 + r + (hi << 3);
        float w = gws[row];
        if (w != 0.0f) {
          atomicAdd(&y[(size_t)toks[row] * D_DIM + dcol], w * (acc[m][t][r] + bias));
        }
      }
    }
  }
}

// ---------------------------------------------------------------------------
// Kernel 3: aux loss = E * sum_e (top1_count_e/N) * (probsum_e/N)
// ---------------------------------------------------------------------------
__global__ void moe_finalize_kernel(const int* __restrict__ dens,
                                    const float* __restrict__ probsum,
                                    float* __restrict__ out_aux) {
  if (threadIdx.x == 0 && blockIdx.x == 0) {
    float s = 0.0f;
#pragma unroll
    for (int e = 0; e < E_NUM; ++e)
      s += ((float)dens[e] / (float)N_TOK) * (probsum[e] / (float)N_TOK);
    out_aux[0] = (float)E_NUM * s;
  }
}

// ---------------------------------------------------------------------------
extern "C" void kernel_launch(void* const* d_in, const int* in_sizes, int n_in,
                              void* d_out, int out_size, void* d_ws, size_t ws_size,
                              hipStream_t stream) {
  const float* x  = (const float*)d_in[0];
  const float* wr = (const float*)d_in[1];
  const float* w1 = (const float*)d_in[2];
  const float* b1 = (const float*)d_in[3];
  const float* w2 = (const float*)d_in[4];
  const float* b2 = (const float*)d_in[5];
  float* y   = (float*)d_out;
  float* aux = y + (size_t)N_TOK * D_DIM;

  char* ws = (char*)d_ws;
  int*   counts  = (int*)ws;               // 8 ints
  int*   dens    = counts + 8;             // 8 ints
  float* probsum = (float*)(counts + 16);  // 8 floats (header = 256 B)
  int*   bucket  = (int*)(ws + 256);                               // E*N ints
  float* gwt     = (float*)(ws + 256 + (size_t)E_NUM * N_TOK * 4); // E*N floats

  // Pair-packed bf16 weights live after the first MiB of workspace
  const size_t npairs = (size_t)E_NUM * D_DIM * F_DIM / 2;  // 16,777,216 each
  unsigned int* w1p = (unsigned int*)(ws + (1u << 20));
  unsigned int* w2p = w1p + npairs;
  const size_t ws_needed = (1u << 20) + 2 * npairs * sizeof(unsigned int);
  const bool packed = ws_size >= ws_needed;  // constant across calls

  size_t total = (size_t)N_TOK * D_DIM;
  moe_zero_kernel<<<(int)((total + 255) / 256), 256, 0, stream>>>(
      y, (unsigned int*)ws);

  moe_router_kernel<<<N_TOK / 8, 256, 0, stream>>>(x, wr, counts, dens, probsum,
                                                   bucket, gwt);

  dim3 grid(N_TOK / TILE_M, E_NUM);
  if (packed) {
    int pg = (int)((npairs + 255) / 256);
    moe_pack_kernel<<<pg, 256, 0, stream>>>(w1, w1p, (int)npairs, F_DIM);
    moe_pack_kernel<<<pg, 256, 0, stream>>>(w2, w2p, (int)npairs, D_DIM);
    moe_ffn_kernel<true><<<grid, 512, 0, stream>>>(
        x, w1, w1p, b1, w2, w2p, b2, counts, bucket, gwt, y);
  } else {
    moe_ffn_kernel<false><<<grid, 512, 0, stream>>>(
        x, w1, w1p, b1, w2, w2p, b2, counts, bucket, gwt, y);
  }

  moe_finalize_kernel<<<1, 32, 0, stream>>>(dens, probsum, aux);
}
